// Decoder_87488483820396
// MI455X (gfx1250) — matrix operand order repaired and checked by reading, not compiled, not run on hardware
//
#include <hip/hip_runtime.h>
#include <hip/hip_bf16.h>

typedef __attribute__((ext_vector_type(16))) _Float16 v16h;
typedef __attribute__((ext_vector_type(8)))  _Float16 v8h;
typedef __attribute__((ext_vector_type(4)))  _Float16 v4h;
typedef __attribute__((ext_vector_type(2)))  _Float16 v2h;
typedef __attribute__((ext_vector_type(8)))  float    v8f;

constexpr int TL  = 6;      // layers
constexpr int TB  = 4;      // batch
constexpr int TS  = 512;    // seq len
constexpr int TD  = 1024;   // model dim
constexpr int TH  = 16;     // heads
constexpr int TDK = 64;     // head dim
constexpr int TFF = 4096;   // ffn dim
constexpr int TM  = TB * TS;         // 2048 activation rows
constexpr int THD = TH * TDK;        // 1024

// ---------------------------------------------------------------- embedding
__global__ __launch_bounds__(256) void embed_kernel(
    const int* __restrict__ ids, const float* __restrict__ tok,
    const float* __restrict__ pos, float* __restrict__ x) {
  const int s = blockIdx.x, b = blockIdx.y;
  const int id = ids[b * TS + s];
  const float* te = tok + (size_t)id * TD;
  const float* pe = pos + (size_t)s * TD;
  float* xr = x + ((size_t)b * TS + s) * TD;
  for (int d = threadIdx.x; d < TD; d += 256) xr[d] = te[d] + pe[d];
}

// ------------------------------------------------------------ tiled f16 GEMM
// C[M,N] = op(A[M,K] @ B[K,N]); A: f32 or f16 acts, B: f32 weights.
// Block tile 64x128, BK=32, double-buffered LDS, software-pipelined global
// prefetch. 8 waves each compute a 16x64 strip (4 WMMA subtiles).
template <typename AT, typename CT, bool RELU>
__global__ __launch_bounds__(256) void gemm_kernel(
    const AT* __restrict__ A, const float* __restrict__ Bm,
    CT* __restrict__ C, int N, int Kd) {
  __shared__ _Float16 As[2][64][40];    // padded rows (80B) -> ds_*_b128 ok
  __shared__ _Float16 Bs[2][128][40];   // Bs[n][k]: K-contiguous per column
  const int tid = threadIdx.x, lane = tid & 31, wave = tid >> 5;
  const int m0 = blockIdx.y * 64, n0 = blockIdx.x * 128;
  const int wm = (wave & 3) * 16, wn = (wave >> 2) * 64;

  // staging assignments
  const int arow = tid >> 2, akc = (tid & 3) * 8;       // A: 64 rows x 32 k
  const int bn8 = (tid & 15) * 8, bk2 = (tid >> 4) * 2; // B: 2 k-rows x 8 n

  const AT*    aptr  = A + (size_t)(m0 + arow) * Kd + akc;
  const float* bptr0 = Bm + (size_t)bk2 * N + n0 + bn8;
  const float* bptr1 = bptr0 + N;

  AT    areg[8];
  float breg0[8], breg1[8];

  // prefetch tile 0 into registers
#pragma unroll
  for (int j = 0; j < 8; ++j) areg[j] = aptr[j];
#pragma unroll
  for (int j = 0; j < 8; ++j) { breg0[j] = bptr0[j]; breg1[j] = bptr1[j]; }

  auto stage = [&](int buf) {
    v8h apack;
#pragma unroll
    for (int j = 0; j < 8; ++j) apack[j] = (_Float16)areg[j];
    *(v8h*)&As[buf][arow][akc] = apack;          // one ds_store_b128
#pragma unroll
    for (int j = 0; j < 8; ++j) {                // packed b32 transpose stores
      v2h p; p[0] = (_Float16)breg0[j]; p[1] = (_Float16)breg1[j];
      *(v2h*)&Bs[buf][bn8 + j][bk2] = p;
    }
  };
  stage(0);

  v8f acc[4] = {};
  const int mm = wm + (lane & 15);
  const int kb = (lane >> 4) * 8, kb2 = (lane >> 4) * 16;

  int buf = 0;
  for (int k0 = 0; k0 < Kd; k0 += 32) {
    __syncthreads();                             // LDS[buf] ready
    const bool more = (k0 + 32) < Kd;
    if (more) {                                  // pipeline next global tile
      const AT* ap = aptr + k0 + 32;
#pragma unroll
      for (int j = 0; j < 8; ++j) areg[j] = ap[j];
      const float* bp0 = bptr0 + (size_t)(k0 + 32) * N;
      const float* bp1 = bp0 + N;
#pragma unroll
      for (int j = 0; j < 8; ++j) { breg0[j] = bp0[j]; breg1[j] = bp1[j]; }
      if (k0 + 64 < Kd)                          // global_prefetch_b8 ahead
        __builtin_prefetch(bptr0 + (size_t)(k0 + 64) * N, 0, 1);
    }
    // A fragment: lane row (lane&15); K = 16*(lane>>4)+{0..15}
    v16h a;
#pragma unroll
    for (int j = 0; j < 8; ++j) {
      a[j]     = As[buf][mm][kb2 + j];
      a[8 + j] = As[buf][mm][kb2 + 8 + j];
    }
#pragma unroll
    for (int st = 0; st < 4; ++st) {
      const int nn = wn + st * 16 + (lane & 15);
      v16h bf;
#pragma unroll
      for (int j = 0; j < 16; ++j) bf[j] = Bs[buf][nn][kb2 + j];
      acc[st] = __builtin_amdgcn_wmma_f32_16x16x32_f16(
          false, a, false, bf, (short)0, acc[st], false, false);
    }
    if (more) stage(buf ^ 1);                    // fill other buffer
    buf ^= 1;
  }

  // C/D layout: lane -> N = lane&15, VGPR r -> M = r + 8*(lane>>4)
  const int mr = (lane >> 4) * 8, ncol = lane & 15;
#pragma unroll
  for (int st = 0; st < 4; ++st)
#pragma unroll
    for (int r = 0; r < 8; ++r) {
      float v = acc[st][r];
      if (RELU) v = fmaxf(v, 0.0f);
      C[(size_t)(m0 + wm + mr + r) * N + (n0 + wn + st * 16 + ncol)] = (CT)v;
    }
}

// ------------------------------------------------- fused attention per (b,h)
// One workgroup: 16 query rows x all 512 keys. scores -> mask -> softmax ->
// probs @ V. Q/K/V are f16 [B*S, H*DK] row-major (head-dim contiguous).
template <bool MASKED, bool WRITE_ATTN>
__global__ __launch_bounds__(256) void attn_kernel(
    const _Float16* __restrict__ Qh, const _Float16* __restrict__ Kh,
    const _Float16* __restrict__ Vh, const int* __restrict__ ids,
    _Float16* __restrict__ Ctx, float* __restrict__ attn_out) {
  __shared__ _Float16 Qs[16][72];
  __shared__ float    Ss[16][516];
  __shared__ _Float16 Vt[64][520];       // V transposed: Vt[d][key]
  __shared__ float    red[16][16];
  __shared__ float    rowstat[16];
  __shared__ float    Cred[4][8][32];

  const int tid = threadIdx.x, lane = tid & 31, wave = tid >> 5;
  const int qblk = blockIdx.x, h = blockIdx.y, b = blockIdx.z;
  const int q0 = qblk * 16;
  const size_t baseBH = (size_t)b * TS * THD + (size_t)h * TDK;

  {  // Q tile: 16 rows x 64, vectorized 4-half copies (one per thread)
    const int i = tid * 4;
    const int r = i >> 6, d = i & 63;
    *(v4h*)&Qs[r][d] = *(const v4h*)(Qh + baseBH + (size_t)(q0 + r) * THD + d);
  }
  {  // V transpose: each thread owns a key pair, packed b32 stores
    const int kp = tid * 2;  // keys kp, kp+1
    const _Float16* vp0 = Vh + baseBH + (size_t)kp * THD;
    const _Float16* vp1 = vp0 + THD;
#pragma unroll
    for (int c = 0; c < 4; ++c) {
      v16h a0 = *(const v16h*)(vp0 + c * 16);
      v16h a1 = *(const v16h*)(vp1 + c * 16);
#pragma unroll
      for (int j = 0; j < 16; ++j) {
        v2h p; p[0] = a0[j]; p[1] = a1[j];
        *(v2h*)&Vt[c * 16 + j][kp] = p;
      }
    }
  }
  __syncthreads();

  // ---- scores = Q @ K^T * 1/sqrt(DK); wave w owns keys [64w, 64w+64)
  {
    v8f sc[4] = {};
    const int mm = lane & 15, kb = (lane >> 4) * 8, k16 = (lane >> 4) * 16;
#pragma unroll
    for (int ks = 0; ks < 64; ks += 32) {
      v16h a;
#pragma unroll
      for (int j = 0; j < 8; ++j) { a[j] = Qs[mm][ks + k16 + j]; a[8 + j] = Qs[mm][ks + k16 + 8 + j]; }
#pragma unroll
      for (int st = 0; st < 4; ++st) {
        const int nk = wave * 64 + st * 16 + (lane & 15);
        // K rows are head-dim contiguous => B fragment is a straight 32B load
        const _Float16* kp = Kh + baseBH + (size_t)nk * THD + ks + k16;
        v16h bf = *(const v16h*)kp;
        sc[st] = __builtin_amdgcn_wmma_f32_16x16x32_f16(
            false, a, false, bf, (short)0, sc[st], false, false);
      }
    }
    const int mr = (lane >> 4) * 8;
#pragma unroll
    for (int st = 0; st < 4; ++st)
#pragma unroll
      for (int r = 0; r < 8; ++r) {
        const int mloc = mr + r;
        const int n = wave * 64 + st * 16 + (lane & 15);
        float v = sc[st][r] * 0.125f;                    // 1/sqrt(64)
        if (MASKED) {
          if ((n > q0 + mloc) || (ids[b * TS + n] == 0)) v = -1e9f;
        }
        Ss[mloc][n] = v;
      }
  }
  __syncthreads();

  // ---- softmax over 512 keys (16 threads per row, 32 cols each)
  {
    const int row = tid >> 4, seg = tid & 15, cb = seg * 32;
    float mx = -3.4e38f;
    for (int j = 0; j < 32; ++j) mx = fmaxf(mx, Ss[row][cb + j]);
    red[row][seg] = mx;
    __syncthreads();
    if (seg == 0) {
      float m2 = red[row][0];
      for (int i = 1; i < 16; ++i) m2 = fmaxf(m2, red[row][i]);
      rowstat[row] = m2;
    }
    __syncthreads();
    const float rm = rowstat[row];
    float ssum = 0.0f;
    for (int j = 0; j < 32; ++j) {
      const float e = __expf(Ss[row][cb + j] - rm);
      Ss[row][cb + j] = e;
      ssum += e;
    }
    red[row][seg] = ssum;
    __syncthreads();
    if (seg == 0) {
      float s2 = 0.0f;
      for (int i = 0; i < 16; ++i) s2 += red[row][i];
      rowstat[row] = s2;
    }
    __syncthreads();
    const float inv = 1.0f / rowstat[row];
    for (int j = 0; j < 32; ++j) {
      const float pv = Ss[row][cb + j] * inv;
      Ss[row][cb + j] = pv;
      if (WRITE_ATTN)
        attn_out[(((size_t)(b * TH + h) * TS) + (q0 + row)) * TS + cb + j] = pv;
    }
  }
  __syncthreads();

  // ---- ctx = P[16x512] @ V[512x64]; wave -> (subtile = w&3, k-parity = w>>2)
  {
    const int sub = wave & 3, par = wave >> 2;
    const int mm = lane & 15, kb = (lane >> 4) * 8, k16 = (lane >> 4) * 16;
    const int dloc = sub * 16 + (lane & 15);
    v8f cc = {};
    for (int kc = par; kc < 16; kc += 2) {
      const int ko = kc * 32;
      v16h a;
#pragma unroll
      for (int j = 0; j < 8; ++j) {
        a[j]     = (_Float16)Ss[mm][ko + k16 + j];
        a[8 + j] = (_Float16)Ss[mm][ko + k16 + 8 + j];
      }
      v16h bf;
#pragma unroll
      for (int j = 0; j < 16; ++j) bf[j] = Vt[dloc][ko + k16 + j];
      cc = __builtin_amdgcn_wmma_f32_16x16x32_f16(
          false, a, false, bf, (short)0, cc, false, false);
    }
    if (par == 1) {
#pragma unroll
      for (int r = 0; r < 8; ++r) Cred[sub][r][lane] = cc[r];
    }
    __syncthreads();
    if (par == 0) {
      const int mr = (lane >> 4) * 8;
#pragma unroll
      for (int r = 0; r < 8; ++r) {
        const float v = cc[r] + Cred[sub][r][lane];
        Ctx[baseBH + (size_t)(q0 + mr + r) * THD + dloc] = (_Float16)v;
      }
    }
  }
}

// ----------------------------------------------- residual add + layernorm
__global__ __launch_bounds__(256) void addln_kernel(
    const float* __restrict__ y, const float* __restrict__ xin,
    const float* __restrict__ g, const float* __restrict__ bb,
    float* __restrict__ xout) {
  __shared__ float sred[256];
  const int row = blockIdx.x, tid = threadIdx.x;
  const float* yr = y + (size_t)row * TD;
  const float* xr = xin + (size_t)row * TD;
  float v4[4];
  float s = 0.0f;
#pragma unroll
  for (int i = 0; i < 4; ++i) {
    const int d = tid + i * 256;
    v4[i] = yr[d] + xr[d];
    s += v4[i];
  }
  sred[tid] = s; __syncthreads();
  for (int off = 128; off > 0; off >>= 1) {
    if (tid < off) sred[tid] += sred[tid + off];
    __syncthreads();
  }
  const float mean = sred[0] / (float)TD;
  __syncthreads();
  float vs = 0.0f;
#pragma unroll
  for (int i = 0; i < 4; ++i) { const float d = v4[i] - mean; vs += d * d; }
  sred[tid] = vs; __syncthreads();
  for (int off = 128; off > 0; off >>= 1) {
    if (tid < off) sred[tid] += sred[tid + off];
    __syncthreads();
  }
  const float inv = rsqrtf(sred[0] / (float)TD + 1e-5f);
  float* xo = xout + (size_t)row * TD;
#pragma unroll
  for (int i = 0; i < 4; ++i) {
    const int d = tid + i * 256;
    xo[d] = (v4[i] - mean) * inv * g[d] + bb[d];
  }
}

// ---------------------------------------------------------------- launcher
extern "C" void kernel_launch(void* const* d_in, const int* in_sizes, int n_in,
                              void* d_out, int out_size, void* d_ws, size_t ws_size,
                              hipStream_t stream) {
  (void)in_sizes; (void)n_in; (void)out_size; (void)ws_size;
  const int*   dec  = (const int*)  d_in[0];
  const float* tok  = (const float*)d_in[1];
  const float* pos  = (const float*)d_in[2];
  const float* Wq1  = (const float*)d_in[3];
  const float* Wk1  = (const float*)d_in[4];
  const float* Wv1  = (const float*)d_in[5];
  const float* Wo1  = (const float*)d_in[6];
  const float* g1   = (const float*)d_in[7];
  const float* b1   = (const float*)d_in[8];
  const float* Wq2  = (const float*)d_in[9];
  const float* Wk2  = (const float*)d_in[10];
  const float* Wv2  = (const float*)d_in[11];
  const float* Wo2  = (const float*)d_in[12];
  const float* g2   = (const float*)d_in[13];
  const float* b2   = (const float*)d_in[14];
  const float* Wff1 = (const float*)d_in[15];
  const float* Wff2 = (const float*)d_in[16];
  const float* gff  = (const float*)d_in[17];
  const float* bff  = (const float*)d_in[18];

  float* out_x    = (float*)d_out;
  float* out_attn = out_x + (size_t)TM * TD;

  char* wp = (char*)d_ws;
  auto take = [&](size_t bytes) -> char* {
    char* r = wp; wp += (bytes + 255) & ~(size_t)255; return r;
  };
  float*    x    = (float*)   take((size_t)TM * TD  * 4);
  float*    y    = (float*)   take((size_t)TM * TD  * 4);
  _Float16* Qh   = (_Float16*)take((size_t)TM * TD  * 2);
  _Float16* Kh   = (_Float16*)take((size_t)TM * TD  * 2);
  _Float16* Vh   = (_Float16*)take((size_t)TM * TD  * 2);
  _Float16* Ctxh = (_Float16*)take((size_t)TM * TD  * 2);
  _Float16* Hh   = (_Float16*)take((size_t)TM * TFF * 2);

  const dim3 blk(256);
  const dim3 gEmb(TS, TB);
  const dim3 gGemmD(TD / 128, TM / 64);    // N = 1024
  const dim3 gGemmF(TFF / 128, TM / 64);   // N = 4096
  const dim3 gAttn(TS / 16, TH, TB);

  embed_kernel<<<gEmb, blk, 0, stream>>>(dec, tok, pos, x);

  for (int l = 0; l < TL; ++l) {
    const size_t wOff  = (size_t)l * TD * TD;
    const size_t fOff1 = (size_t)l * TD * TFF;
    const size_t fOff2 = (size_t)l * TFF * TD;
    const size_t vOff  = (size_t)l * TD;

    // ---- MHA1 (masked)
    gemm_kernel<float, _Float16, false><<<gGemmD, blk, 0, stream>>>(x, Wq1 + wOff, Qh, TD, TD);
    gemm_kernel<float, _Float16, false><<<gGemmD, blk, 0, stream>>>(x, Wk1 + wOff, Kh, TD, TD);
    gemm_kernel<float, _Float16, false><<<gGemmD, blk, 0, stream>>>(x, Wv1 + wOff, Vh, TD, TD);
    attn_kernel<true, false><<<gAttn, blk, 0, stream>>>(Qh, Kh, Vh, dec, Ctxh, nullptr);
    gemm_kernel<_Float16, float, false><<<gGemmD, blk, 0, stream>>>(Ctxh, Wo1 + wOff, y, TD, TD);
    addln_kernel<<<TM, blk, 0, stream>>>(y, x, g1 + vOff, b1 + vOff, x);

    // ---- MHA2 (unmasked, attention probs written out)
    gemm_kernel<float, _Float16, false><<<gGemmD, blk, 0, stream>>>(x, Wq2 + wOff, Qh, TD, TD);
    gemm_kernel<float, _Float16, false><<<gGemmD, blk, 0, stream>>>(x, Wk2 + wOff, Kh, TD, TD);
    gemm_kernel<float, _Float16, false><<<gGemmD, blk, 0, stream>>>(x, Wv2 + wOff, Vh, TD, TD);
    attn_kernel<false, true><<<gAttn, blk, 0, stream>>>(
        Qh, Kh, Vh, dec, Ctxh, out_attn + (size_t)l * TB * TH * TS * TS);
    gemm_kernel<_Float16, float, false><<<gGemmD, blk, 0, stream>>>(Ctxh, Wo2 + wOff, y, TD, TD);
    addln_kernel<<<TM, blk, 0, stream>>>(y, x, g2 + vOff, b2 + vOff, x);

    // ---- FFN
    gemm_kernel<float, _Float16, true ><<<gGemmF, blk, 0, stream>>>(x, Wff1 + fOff1, Hh, TFF, TD);
    gemm_kernel<_Float16, float, false><<<gGemmD, blk, 0, stream>>>(Hh, Wff2 + fOff2, y, TD, TFF);
    addln_kernel<<<TM, blk, 0, stream>>>(y, x, gff + vOff, bff + vOff, x);
  }

  hipMemcpyAsync(d_out, x, (size_t)TM * TD * sizeof(float),
                 hipMemcpyDeviceToDevice, stream);
}
